// EncoderLayer_42477226557473
// MI455X (gfx1250) — compile-verified, matrix operand order checked by reading, not run on hardware
//
#include <hip/hip_runtime.h>

// ---------------- CDNA5 (gfx1250) types ----------------
typedef __bf16 bf16;
typedef __bf16 v16bf __attribute__((ext_vector_type(16)));
typedef float  v8f   __attribute__((ext_vector_type(8)));
typedef int    i32x4 __attribute__((ext_vector_type(4)));
typedef __attribute__((address_space(1))) i32x4 gi32x4;   // global
typedef __attribute__((address_space(3))) i32x4 li32x4;   // LDS

#define B_    2
#define S_    4096
#define D_    1024
#define H_    16
#define HL_   14          // local (sliding-window) heads
#define DH_   64
#define W_    128
#define DFF_  4096
#define M_    (B_ * S_)   // 8192 rows

// ---------------- CDNA5 async LDS-DMA path (guarded) ----------------
#ifndef __has_builtin
#define __has_builtin(x) 0
#endif
#if __has_builtin(__builtin_amdgcn_global_load_async_to_lds_b128)
#define HAVE_ASYNC_LDS 1
#else
#define HAVE_ASYNC_LDS 0
#endif

// Stage one 16-byte chunk global(bf16) -> LDS. Async (ASYNCcnt) when available.
__device__ __forceinline__ void stage16(const bf16* g, bf16* l) {
#if HAVE_ASYNC_LDS
  __builtin_amdgcn_global_load_async_to_lds_b128((gi32x4*)g, (li32x4*)l, 0, 0);
#else
  *(uint4*)l = *(const uint4*)g;
#endif
}

__device__ __forceinline__ void async_wait0() {
#if HAVE_ASYNC_LDS
#if __has_builtin(__builtin_amdgcn_s_wait_asynccnt)
  __builtin_amdgcn_s_wait_asynccnt(0);
#else
  asm volatile("s_wait_asynccnt 0" ::: "memory");
#endif
#endif
}

// ---------------- WMMA helpers ----------------
__device__ __forceinline__ v8f wmma_bf16(v16bf a, v16bf b, v8f c) {
  return __builtin_amdgcn_wmma_f32_16x16x32_bf16(false, a, false, b, (short)0, c,
                                                 false, false);
}

// Load a 16x32 bf16 operand fragment from LDS or global. base[i*ld + k], i the
// 16-wide dim, k the 32-wide K dim. Lanes 0-15 -> i=lane, K {0..7,16..23};
// lanes 16-31 -> i=lane-16, K {8..15,24..31}. Two 16B loads per lane
// (base 16B-aligned, ld*2 % 16 == 0).
__device__ __forceinline__ v16bf load_frag16(const bf16* base, int ld, int lane) {
  const int half = (lane >> 4) & 1;
  const int m    = lane & 15;
  union { v16bf v; uint4 q[2]; } f;
  const bf16* p = base + m * ld + half * 8;
  f.q[0] = *(const uint4*)(p);
  f.q[1] = *(const uint4*)(p + 16);
  return f.v;
}

// ---------------- One-time precision staging ----------------
__global__ void __launch_bounds__(256)
cvt_bf16_kernel(const float* __restrict__ src, bf16* __restrict__ dst) {
  size_t i = ((size_t)blockIdx.x * 256 + threadIdx.x) * 4;
  float4 v = *(const float4*)(src + i);
  union { bf16 e[4]; uint2 u; } t;
  t.e[0] = (bf16)v.x; t.e[1] = (bf16)v.y; t.e[2] = (bf16)v.z; t.e[3] = (bf16)v.w;
  *(uint2*)(dst + i) = t.u;
}

// W[K][N] f32 -> Wt[N][K] bf16 (32x32 LDS tile transpose).
__global__ void __launch_bounds__(256)
cvt_tr_kernel(const float* __restrict__ W, bf16* __restrict__ Wt, int K, int N) {
  __shared__ bf16 t[32][33];
  const int n0 = blockIdx.x * 32, k0 = blockIdx.y * 32;
  const int tx = threadIdx.x & 31, ty = threadIdx.x >> 5;   // ty 0..7
#pragma unroll
  for (int j = 0; j < 4; ++j) {
    int k = k0 + ty + j * 8;
    t[tx][ty + j * 8] = (bf16)W[(size_t)k * N + n0 + tx];   // t[n_loc][k_loc]
  }
  __syncthreads();
#pragma unroll
  for (int j = 0; j < 4; ++j) {
    int n = n0 + ty + j * 8;
    Wt[(size_t)n * K + k0 + tx] = t[ty + j * 8][tx];
  }
}

// ---------------- GEMM: C = act(A[M,K]bf16 @ Wt[N,K]bf16^T + bias) ----------------
#define GM  128
#define GN  128
#define GK  64
#define LDA 72   // bf16 elems; 144B rows: 16B-aligned, staggers LDS banks
#define LDB 72

template <int ACT, int OUTBF>  // ACT: 0 none, 1 leaky(0.2). OUTBF: 1 = bf16 out
__global__ void __launch_bounds__(256)
gemm_bias_kernel(const bf16* __restrict__ A, const bf16* __restrict__ Wt,
                 const float* __restrict__ bias, void* __restrict__ Cv,
                 int M, int N, int K) {
  // Double-buffered: async DMA for tile kt+1 overlaps WMMA on tile kt.
  __shared__ alignas(16) bf16 As[2][GM * LDA];   // [buf][row][k]
  __shared__ alignas(16) bf16 Bs[2][GN * LDB];   // [buf][n][k]

  const int tid  = threadIdx.x;
  const int lane = tid & 31;
  const int wid  = tid >> 5;
  const int wm   = wid & 3;       // 4 waves along M
  const int wn   = wid >> 2;      // 2 waves along N
  const int bm0  = blockIdx.y * GM;
  const int bn0  = blockIdx.x * GN;

  v8f acc[2][4];
#pragma unroll
  for (int mi = 0; mi < 2; ++mi)
#pragma unroll
    for (int ni = 0; ni < 4; ++ni)
      acc[mi][ni] = (v8f){0.f, 0.f, 0.f, 0.f, 0.f, 0.f, 0.f, 0.f};

  const int ksteps = K / GK;

  // Prologue: stage tile 0 into buffer 0.
#pragma unroll
  for (int i = 0; i < 4; ++i) {
    int c = tid + 256 * i, r = c >> 3, c8 = c & 7;
    stage16(A  + (size_t)(bm0 + r) * K + c8 * 8, &As[0][r * LDA + c8 * 8]);
    stage16(Wt + (size_t)(bn0 + r) * K + c8 * 8, &Bs[0][r * LDB + c8 * 8]);
  }

  for (int kt = 0; kt < ksteps; ++kt) {
    const int cur = kt & 1;
    async_wait0();        // own DMAs (issued last iteration) done
    __syncthreads();      // everyone's DMAs visible; frees the other buffer
    if (kt + 1 < ksteps) {
      const int nxt = cur ^ 1;
      const int k0  = (kt + 1) * GK;
#pragma unroll
      for (int i = 0; i < 4; ++i) {
        int c = tid + 256 * i, r = c >> 3, c8 = c & 7;
        stage16(A  + (size_t)(bm0 + r) * K + k0 + c8 * 8, &As[nxt][r * LDA + c8 * 8]);
        stage16(Wt + (size_t)(bn0 + r) * K + k0 + c8 * 8, &Bs[nxt][r * LDB + c8 * 8]);
      }
      if (kt + 2 < ksteps) {  // pull tile kt+2 toward GL2
        __builtin_prefetch((const void*)(A  + (size_t)(bm0 + (tid >> 1)) * K + k0 + GK), 0, 1);
        __builtin_prefetch((const void*)(Wt + (size_t)(bn0 + (tid >> 1)) * K + k0 + GK), 0, 1);
      }
    }
#pragma unroll
    for (int ks = 0; ks < 2; ++ks) {
      v16bf a0 = load_frag16(&As[cur][(wm * 32 + 0)  * LDA + ks * 32], LDA, lane);
      v16bf a1 = load_frag16(&As[cur][(wm * 32 + 16) * LDA + ks * 32], LDA, lane);
#pragma unroll
      for (int ni = 0; ni < 4; ++ni) {
        v16bf b = load_frag16(&Bs[cur][(wn * 64 + ni * 16) * LDB + ks * 32], LDB, lane);
        acc[0][ni] = wmma_bf16(a0, b, acc[0][ni]);
        acc[1][ni] = wmma_bf16(a1, b, acc[1][ni]);
      }
    }
  }

  // Epilogue. C layout: VGPR r -> M = half*8 + r; N = lane&15.
  const int half = (lane >> 4) & 1, ln = lane & 15;
#pragma unroll
  for (int mi = 0; mi < 2; ++mi)
#pragma unroll
    for (int ni = 0; ni < 4; ++ni) {
      int gn = bn0 + wn * 64 + ni * 16 + ln;
      float bv = bias[gn];
#pragma unroll
      for (int r = 0; r < 8; ++r) {
        int gm  = bm0 + wm * 32 + mi * 16 + half * 8 + r;
        float v = acc[mi][ni][r] + bv;
        if (ACT == 1) v = (v > 0.f) ? v : 0.2f * v;
        if (OUTBF) ((bf16*)Cv)[(size_t)gm * N + gn] = (bf16)v;
        else       ((float*)Cv)[(size_t)gm * N + gn] = v;
      }
    }
}

// ---------------- Attention (flash-style, band-local or global heads) ----------------
#define LDK  72    // Kc[key][feat]
#define LDVT 136   // Vt[feat][key]
#define LDP  72    // per-wave P half tile [16][64]

__global__ void __launch_bounds__(256)
attn_kernel(const bf16* __restrict__ Qb, const bf16* __restrict__ Kb,
            const bf16* __restrict__ Vb, const float* __restrict__ Mask,
            bf16* __restrict__ O) {
  __shared__ alignas(16) bf16 Kc[2][128 * LDK];
  __shared__ alignas(16) bf16 Vt[2][64 * LDVT];
  __shared__ alignas(16) bf16 Ps[8 * 16 * LDP];

  const int tid = threadIdx.x, lane = tid & 31, wid = tid >> 5;
  const int half = (lane >> 4) & 1, ln = lane & 15;
  const int qc = blockIdx.x;
  const int bh = blockIdx.y;
  const int b = bh / H_, h = bh % H_;
  const bool local = (h < HL_);
  const int q0 = qc * 128 + wid * 16;
  const float scale = 0.125f;   // applied to fp32 scores (1/sqrt(64))

  // Q fragments straight from global bf16 (global_load_b128 pairs).
  v16bf qf[2];
  {
    const bf16* qbase = Qb + ((size_t)b * S_ + q0) * D_ + h * DH_;
    qf[0] = load_frag16(qbase, D_, lane);
    qf[1] = load_frag16(qbase + 32, D_, lane);
  }

  v8f out[4];
#pragma unroll
  for (int dt = 0; dt < 4; ++dt) out[dt] = (v8f){0.f,0.f,0.f,0.f,0.f,0.f,0.f,0.f};
  float mrow[8], lrow[8];
#pragma unroll
  for (int r = 0; r < 8; ++r) { mrow[r] = -1e30f; lrow[r] = 0.f; }

  const int kc_lo = local ? (qc > 0 ? qc - 1 : 0) : 0;
  const int kc_hi = local ? (qc < 31 ? qc + 1 : 31) : 31;
  const int nkc   = kc_hi - kc_lo + 1;

  // Stage key chunk kc into buffer buf: K via async DMA (key-major), V via
  // vector load + transposed LDS stores (feat-major).
  auto stage_kv = [&](int buf, int kc) {
    const bf16* kb = Kb + ((size_t)b * S_ + kc * 128) * D_ + h * DH_;
    const bf16* vb = Vb + ((size_t)b * S_ + kc * 128) * D_ + h * DH_;
#pragma unroll
    for (int i = 0; i < 4; ++i) {
      int c = tid + 256 * i, r = c >> 3, c8 = c & 7;
      stage16(kb + (size_t)r * D_ + c8 * 8, &Kc[buf][r * LDK + c8 * 8]);
    }
#pragma unroll
    for (int i = 0; i < 4; ++i) {
      int c = tid + 256 * i, r = c >> 3, c8 = c & 7;
      union { uint4 u; bf16 e[8]; } t;
      t.u = *(const uint4*)(vb + (size_t)r * D_ + c8 * 8);
      int f0 = c8 * 8;
#pragma unroll
      for (int j = 0; j < 8; ++j) Vt[buf][(f0 + j) * LDVT + r] = t.e[j];
    }
  };

  stage_kv(0, kc_lo);

  for (int ii = 0; ii < nkc; ++ii) {
    const int kc  = kc_lo + ii;
    const int cur = ii & 1;
    async_wait0();
    __syncthreads();
    if (ii + 1 < nkc) stage_kv(cur ^ 1, kc + 1);   // overlap with compute

    // Scores: 16x128 = 8 N-tiles, K=64 -> 2 WMMA each.
    v8f st[8];
#pragma unroll
    for (int nt = 0; nt < 8; ++nt) {
      st[nt] = (v8f){0.f,0.f,0.f,0.f,0.f,0.f,0.f,0.f};
#pragma unroll
      for (int kk = 0; kk < 2; ++kk) {
        v16bf kf = load_frag16(&Kc[cur][(nt * 16) * LDK + kk * 32], LDK, lane);
        st[nt] = wmma_bf16(qf[kk], kf, st[nt]);
      }
    }

    // Scale + mask + row max.
    float rowmax[8];
#pragma unroll
    for (int r = 0; r < 8; ++r) rowmax[r] = -1e30f;
#pragma unroll
    for (int nt = 0; nt < 8; ++nt) {
      int jk = kc * 128 + nt * 16 + ln;
#pragma unroll
      for (int r = 0; r < 8; ++r) {
        int iq = q0 + half * 8 + r;
        float s = st[nt][r] * scale;
        if (local) {
          int dd = iq - jk;
          if (dd > W_ || dd < -W_) s = -1e30f;     // band |i-j| <= w
        } else {
          s += Mask[(size_t)b * S_ + jk] * (-1000.0f);
        }
        st[nt][r] = s;
        rowmax[r] = fmaxf(rowmax[r], s);
      }
    }
#pragma unroll
    for (int r = 0; r < 8; ++r) {
      float v = rowmax[r];
      v = fmaxf(v, __shfl_xor(v, 1, 32));
      v = fmaxf(v, __shfl_xor(v, 2, 32));
      v = fmaxf(v, __shfl_xor(v, 4, 32));
      v = fmaxf(v, __shfl_xor(v, 8, 32));
      rowmax[r] = v;
    }

    // Online softmax update.
    float alpha[8], rsum[8];
#pragma unroll
    for (int r = 0; r < 8; ++r) {
      float mn = fmaxf(mrow[r], rowmax[r]);
      alpha[r] = __expf(mrow[r] - mn);
      mrow[r]  = mn;
      rsum[r]  = 0.f;
    }
#pragma unroll
    for (int nt = 0; nt < 8; ++nt)
#pragma unroll
      for (int r = 0; r < 8; ++r) {
        float p = __expf(st[nt][r] - mrow[r]);
        st[nt][r] = p;
        rsum[r] += p;
      }
#pragma unroll
    for (int r = 0; r < 8; ++r) {
      float v = rsum[r];
      v += __shfl_xor(v, 1, 32);
      v += __shfl_xor(v, 2, 32);
      v += __shfl_xor(v, 4, 32);
      v += __shfl_xor(v, 8, 32);
      lrow[r] = lrow[r] * alpha[r] + v;
    }
#pragma unroll
    for (int dt = 0; dt < 4; ++dt)
#pragma unroll
      for (int r = 0; r < 8; ++r) out[dt][r] *= alpha[r];

    // P @ V in two key halves via per-wave LDS P tile (same-wave DS ordering).
    bf16* Pw = &Ps[wid * 16 * LDP];
#pragma unroll
    for (int hh = 0; hh < 2; ++hh) {
#pragma unroll
      for (int t = 0; t < 4; ++t) {
        int nt = hh * 4 + t;
#pragma unroll
        for (int r = 0; r < 8; ++r)
          Pw[(half * 8 + r) * LDP + t * 16 + ln] = (bf16)st[nt][r];
      }
#pragma unroll
      for (int kk2 = 0; kk2 < 2; ++kk2) {
        v16bf pa = load_frag16(&Pw[kk2 * 32], LDP, lane);
#pragma unroll
        for (int dt = 0; dt < 4; ++dt) {
          v16bf vf = load_frag16(&Vt[cur][(dt * 16) * LDVT + hh * 64 + kk2 * 32], LDVT, lane);
          out[dt] = wmma_bf16(pa, vf, out[dt]);
        }
      }
    }
  }

  // Normalize and write bf16 16x64 tile into this head's columns.
  bf16* ob = O + ((size_t)b * S_ + q0) * D_ + h * DH_;
#pragma unroll
  for (int r = 0; r < 8; ++r) {
    float inv = 1.0f / lrow[r];
    int m = half * 8 + r;
#pragma unroll
    for (int dt = 0; dt < 4; ++dt)
      ob[(size_t)m * D_ + dt * 16 + ln] = (bf16)(out[dt][r] * inv);
  }
}

// ---------------- Residual + LayerNorm (optional bf16 side output) ----------------
__global__ void __launch_bounds__(256)
add_ln_kernel(const float* __restrict__ X, const float* __restrict__ Y,
              const float* __restrict__ G, const float* __restrict__ Bt,
              float* __restrict__ O, bf16* __restrict__ Ob) {
  __shared__ float red[16];
  const int row = blockIdx.x;
  const float* x = X + (size_t)row * D_;
  const float* y = Y + (size_t)row * D_;
  float v[4], s = 0.f, s2 = 0.f;
#pragma unroll
  for (int i = 0; i < 4; ++i) {
    int c = threadIdx.x + 256 * i;
    float t = x[c] + y[c];
    v[i] = t; s += t; s2 += t * t;
  }
#pragma unroll
  for (int off = 1; off < 32; off <<= 1) {
    s  += __shfl_xor(s,  off, 32);
    s2 += __shfl_xor(s2, off, 32);
  }
  const int wid = threadIdx.x >> 5, lane = threadIdx.x & 31;
  if (lane == 0) { red[wid] = s; red[8 + wid] = s2; }
  __syncthreads();
  s = 0.f; s2 = 0.f;
#pragma unroll
  for (int w = 0; w < 8; ++w) { s += red[w]; s2 += red[8 + w]; }
  float mu   = s * (1.0f / D_);
  float var  = s2 * (1.0f / D_) - mu * mu;
  float rstd = rsqrtf(var + 1e-6f);
#pragma unroll
  for (int i = 0; i < 4; ++i) {
    int c = threadIdx.x + 256 * i;
    float r = (v[i] - mu) * rstd * G[c] + Bt[c];
    O[(size_t)row * D_ + c] = r;
    if (Ob) Ob[(size_t)row * D_ + c] = (bf16)r;
  }
}

// ---------------- Host-side orchestration ----------------
extern "C" void kernel_launch(void* const* d_in, const int* in_sizes, int n_in,
                              void* d_out, int out_size, void* d_ws, size_t ws_size,
                              hipStream_t stream) {
  (void)in_sizes; (void)n_in; (void)out_size; (void)ws_size;
  const float* x    = (const float*)d_in[0];
  const float* mask = (const float*)d_in[1];
  const float* wq = (const float*)d_in[2];   const float* bq = (const float*)d_in[3];
  const float* wk = (const float*)d_in[4];   const float* bk = (const float*)d_in[5];
  const float* wv = (const float*)d_in[6];   const float* bv = (const float*)d_in[7];
  const float* wo = (const float*)d_in[8];   const float* bo = (const float*)d_in[9];
  const float* w1 = (const float*)d_in[10];  const float* b1 = (const float*)d_in[11];
  const float* w2 = (const float*)d_in[12];  const float* b2 = (const float*)d_in[13];
  const float* g1 = (const float*)d_in[14];  const float* be1 = (const float*)d_in[15];
  const float* g2 = (const float*)d_in[16];  const float* be2 = (const float*)d_in[17];

  // Workspace layout (bump allocator, 256B aligned).
  char* p = (char*)d_ws;
  auto alloc = [&](size_t bytes) -> char* {
    char* r = p; p += (bytes + 255) & ~(size_t)255; return r;
  };
  const size_t MD = (size_t)M_ * D_;
  bf16* xb   = (bf16*)alloc(MD * 2);                 // x in bf16
  bf16* wqt  = (bf16*)alloc((size_t)D_ * D_ * 2);    // transposed bf16 weights
  bf16* wkt  = (bf16*)alloc((size_t)D_ * D_ * 2);
  bf16* wvt  = (bf16*)alloc((size_t)D_ * D_ * 2);
  bf16* wot  = (bf16*)alloc((size_t)D_ * D_ * 2);
  bf16* w1t  = (bf16*)alloc((size_t)D_ * DFF_ * 2);
  bf16* w2t  = (bf16*)alloc((size_t)DFF_ * D_ * 2);
  bf16* qb   = (bf16*)alloc(MD * 2);                 // qb..attb contiguous 64MB
  bf16* kb   = (bf16*)alloc(MD * 2);
  bf16* vb   = (bf16*)alloc(MD * 2);
  bf16* attb = (bf16*)alloc(MD * 2);
  float* t1f = (float*)alloc(MD * 4);                // wo-out, later ffn2-out
  float* o1f = (float*)alloc(MD * 4);                // out1 (f32 residual)
  bf16* o1b  = (bf16*)alloc(MD * 2);                 // out1 (bf16 for FFN1)
  bf16* hbuf = qb;                                   // FFN hidden reuses q/k/v/att

  dim3 blk(256);
  // One-time precision staging.
  cvt_bf16_kernel<<<dim3(MD / 1024), blk, 0, stream>>>(x, xb);
  cvt_tr_kernel<<<dim3(D_ / 32, D_ / 32), blk, 0, stream>>>(wq, wqt, D_, D_);
  cvt_tr_kernel<<<dim3(D_ / 32, D_ / 32), blk, 0, stream>>>(wk, wkt, D_, D_);
  cvt_tr_kernel<<<dim3(D_ / 32, D_ / 32), blk, 0, stream>>>(wv, wvt, D_, D_);
  cvt_tr_kernel<<<dim3(D_ / 32, D_ / 32), blk, 0, stream>>>(wo, wot, D_, D_);
  cvt_tr_kernel<<<dim3(DFF_ / 32, D_ / 32), blk, 0, stream>>>(w1, w1t, D_, DFF_);
  cvt_tr_kernel<<<dim3(D_ / 32, DFF_ / 32), blk, 0, stream>>>(w2, w2t, DFF_, D_);

  dim3 gP(D_ / GN, M_ / GM);
  // Q/K/V projections (bf16 out, consumed by attention).
  gemm_bias_kernel<0, 1><<<gP, blk, 0, stream>>>(xb, wqt, bq, qb, M_, D_, D_);
  gemm_bias_kernel<0, 1><<<gP, blk, 0, stream>>>(xb, wkt, bk, kb, M_, D_, D_);
  gemm_bias_kernel<0, 1><<<gP, blk, 0, stream>>>(xb, wvt, bv, vb, M_, D_, D_);
  // Attention (14 band heads + 2 global heads), bf16 out.
  attn_kernel<<<dim3(S_ / 128, B_ * H_), blk, 0, stream>>>(qb, kb, vb, mask, attb);
  // Output projection (f32 out) + residual LN (f32 + bf16 copies).
  gemm_bias_kernel<0, 0><<<gP, blk, 0, stream>>>(attb, wot, bo, t1f, M_, D_, D_);
  add_ln_kernel<<<dim3(M_), blk, 0, stream>>>(x, t1f, g1, be1, o1f, o1b);
  // FFN: leaky(o1 @ w1 + b1) bf16 -> @ w2 + b2 f32 -> LN.
  gemm_bias_kernel<1, 1><<<dim3(DFF_ / GN, M_ / GM), blk, 0, stream>>>(o1b, w1t, b1, hbuf, M_, DFF_, D_);
  gemm_bias_kernel<0, 0><<<gP, blk, 0, stream>>>(hbuf, w2t, b2, t1f, M_, D_, DFF_);
  add_ln_kernel<<<dim3(M_), blk, 0, stream>>>(o1f, t1f, g2, be2, (float*)d_out, (bf16*)nullptr);
}
